// TransformerBlock_74560632259195
// MI455X (gfx1250) — matrix operand order repaired and checked by reading, not compiled, not run on hardware
//
#include <hip/hip_runtime.h>
#include <hip/hip_bf16.h>

typedef _Float16 f16;
typedef __attribute__((ext_vector_type(16))) _Float16 v16h;
typedef __attribute__((ext_vector_type(8)))  float    v8f;
typedef __attribute__((ext_vector_type(4)))  unsigned int uq4;

#define D_MODEL 2048
#define D_FF    8192
#define NTOK    16384   // B*S
#define SEQ     128
#define NH      16
#define DK      128
#define REL_T   512

// ---------------------------------------------------------------------------
// WMMA helpers (CDNA5 wave32, v_wmma_f32_16x16x32_f16)
// ---------------------------------------------------------------------------
__device__ __forceinline__ v8f wmma_f16(v16h a, v16h b, v8f c) {
  return __builtin_amdgcn_wmma_f32_16x16x32_f16(false, a, false, b, (short)0, c,
                                                false, false);
}

// A fragment: 16x32 f16 (M x K). lanes 0-15 hold K={0..7,16..23},
// lanes 16-31 hold K={8..15,24..31}; M = lane%16.
__device__ __forceinline__ v16h load_frag_a(const f16* base, int ldk, int lane) {
  const int half_id = lane >> 4;
  const int r = lane & 15;
  const f16* p0 = base + r * ldk + half_id * 8;
  const f16* p1 = base + r * ldk + 16 + half_id * 8;
  union { v16h v; struct { uq4 a, b; } q; } u;
  u.q.a = *(const uq4*)p0;
  u.q.b = *(const uq4*)p1;
  return u.v;
}

// B fragment: 32x16 f16 (K x N) from a [N][K]-layout tile (K contiguous).
// lanes 0-15 hold K={0..7,16..23}, lanes 16-31 hold K={8..15,24..31}; N = lane%16.
__device__ __forceinline__ v16h load_frag_b(const f16* base, int ldk, int lane) {
  const int half_id = lane >> 4;
  const int c = lane & 15;
  const f16* p = base + c * ldk + half_id * 8;
  union { v16h v; struct { uq4 a, b; } q; } u;
  u.q.a = *(const uq4*)p;
  u.q.b = *(const uq4*)(p + 16);
  return u.v;
}

// CDNA5 async global -> LDS copy (16B per lane), tracked by ASYNCcnt.
// LDS byte address == low 32 bits of the generic (LDS-aperture) address.
__device__ __forceinline__ void async_b128(f16* lds, const f16* g) {
  unsigned la = (unsigned)(size_t)lds;
  asm volatile("global_load_async_to_lds_b128 %0, %1, off"
               :: "v"(la), "v"(g) : "memory");
}
__device__ __forceinline__ void wait_async0() {
  asm volatile("s_wait_asynccnt 0x0" ::: "memory");
}

// ---------------------------------------------------------------------------
// Convert / transpose prep kernels
// ---------------------------------------------------------------------------
__global__ void cvt_f32_to_f16_kernel(const float* __restrict__ in,
                                      f16* __restrict__ out, size_t n) {
  size_t i = (size_t)blockIdx.x * blockDim.x + threadIdx.x;
  size_t stride = (size_t)gridDim.x * blockDim.x;
  for (; i < n; i += stride) out[i] = (f16)in[i];
}

// W: [K][N] f32 row-major  ->  Wt: [N][K] f16  (tiled 32x32 via LDS)
__global__ __launch_bounds__(256) void transpose_cvt_kernel(
    const float* __restrict__ W, f16* __restrict__ Wt, int K, int N) {
  __shared__ float t[32][33];
  const int n0 = blockIdx.x * 32;
  const int k0 = blockIdx.y * 32;
  const int tx = threadIdx.x & 31;
  const int ty = threadIdx.x >> 5;  // 0..7
#pragma unroll
  for (int i = 0; i < 32; i += 8)
    t[ty + i][tx] = W[(size_t)(k0 + ty + i) * N + n0 + tx];
  __syncthreads();
#pragma unroll
  for (int i = 0; i < 32; i += 8)
    Wt[(size_t)(n0 + ty + i) * K + k0 + tx] = (f16)t[tx][ty + i];
}

// ---------------------------------------------------------------------------
// Tiled WMMA GEMM: C[M,N] = A[M,K] @ B (B given transposed as Bt[N][K]).
// Block tile 128x256x32, 8 waves (2x4), wave tile 64x64 (4x4 of 16x16).
// Double-buffered LDS fed by GLOBAL_LOAD_ASYNC_TO_LDS_B128 (ASYNCcnt).
// EPI: 1 = +bias -> f16 out, 2 = +bias -> f32 out, 3 = +bias, GELU -> f16 out.
// ---------------------------------------------------------------------------
template <int EPI>
__global__ __launch_bounds__(256) void gemm_wmma_kernel(
    const f16* __restrict__ A, const f16* __restrict__ Bt,
    const float* __restrict__ bias, float* __restrict__ Cf,
    f16* __restrict__ Ch, int M, int N, int K) {
  constexpr int BM = 128, BN = 256, BK = 32;
  __shared__ f16 As[2][BM * BK];  // 8KB  x2
  __shared__ f16 Bs[2][BN * BK];  // 16KB x2

  const int tid  = threadIdx.x;
  const int lane = tid & 31;
  const int wave = tid >> 5;
  const int wm   = wave >> 2;  // 0..1
  const int wn   = wave & 3;   // 0..3
  const int m0   = blockIdx.y * BM;
  const int n0   = blockIdx.x * BN;

  v8f acc[4][4];
#pragma unroll
  for (int mt = 0; mt < 4; ++mt)
#pragma unroll
    for (int nt = 0; nt < 4; ++nt) {
      v8f z = {0.f, 0.f, 0.f, 0.f, 0.f, 0.f, 0.f, 0.f};
      acc[mt][nt] = z;
    }

  // issue async copies for one K-tile into buffer `buf`
  auto issue_tile = [&](int buf, int k0) {
#pragma unroll
    for (int i = 0; i < 2; ++i) {  // A: 512 chunks of 8 halves
      const int c  = tid + 256 * i;
      const int r  = c >> 2;
      const int cc = c & 3;
      async_b128(&As[buf][r * BK + cc * 8],
                 &A[(size_t)(m0 + r) * K + k0 + cc * 8]);
    }
#pragma unroll
    for (int i = 0; i < 4; ++i) {  // B: 1024 chunks of 8 halves
      const int c  = tid + 256 * i;
      const int r  = c >> 2;
      const int cc = c & 3;
      async_b128(&Bs[buf][r * BK + cc * 8],
                 &Bt[(size_t)(n0 + r) * K + k0 + cc * 8]);
    }
  };

  const int nk = K / BK;
  issue_tile(0, 0);

  for (int kb = 0; kb < nk; ++kb) {
    const int cur = kb & 1;
    wait_async0();      // our copies into `cur` are done
    __syncthreads();    // everyone's copies done; prev compute on cur^1 done
    if (kb + 1 < nk) issue_tile(cur ^ 1, (kb + 1) * BK);

    v16h af[4];
#pragma unroll
    for (int mt = 0; mt < 4; ++mt)
      af[mt] = load_frag_a(&As[cur][(wm * 64 + mt * 16) * BK], BK, lane);
#pragma unroll
    for (int nt = 0; nt < 4; ++nt) {
      const v16h bf = load_frag_b(&Bs[cur][(wn * 64 + nt * 16) * BK], BK, lane);
#pragma unroll
      for (int mt = 0; mt < 4; ++mt)
        acc[mt][nt] = wmma_f16(af[mt], bf, acc[mt][nt]);
    }
    __syncthreads();    // done reading `cur` before it is refilled
  }

  // Epilogue. C layout: VGPR r -> M = r (+8 for lanes 16-31), N = lane%16.
  const int rbase = m0 + wm * 64 + ((lane >> 4) ? 8 : 0);
  const int cbase = n0 + wn * 64 + (lane & 15);
#pragma unroll
  for (int mt = 0; mt < 4; ++mt)
#pragma unroll
    for (int nt = 0; nt < 4; ++nt) {
      const int col = cbase + nt * 16;
      const float bb = bias[col];
#pragma unroll
      for (int r = 0; r < 8; ++r) {
        const int row = rbase + mt * 16 + r;
        float v = acc[mt][nt][r] + bb;
        if (EPI == 3) v = 0.5f * v * (1.f + erff(v * 0.70710678118654752f));
        if (EPI == 2)
          Cf[(size_t)row * N + col] = v;
        else
          Ch[(size_t)row * N + col] = (f16)v;
      }
    }
}

// ---------------------------------------------------------------------------
// Fused attention per (b,h): scores = QK^T/sqrt(dk) + rel bias, softmax,
// ctx = attn @ V.  S = DK = 128 so every tile is 128x128.
// Per the reference's broadcast: bias(b,i,j) = rel_emb[(b - i) mod 512][j].
// ---------------------------------------------------------------------------
__global__ __launch_bounds__(256) void attn_kernel(
    const f16* __restrict__ Q, const f16* __restrict__ Kmat,
    const f16* __restrict__ V, const float* __restrict__ rel,
    f16* __restrict__ ctx) {
  __shared__ f16 smA[128 * 128];  // Q, then attn probs (f16)
  __shared__ f16 smB[128 * 128];  // K (natural [n][k] layout), then V^T

  const int tid  = threadIdx.x;
  const int lane = tid & 31;
  const int wave = tid >> 5;
  const int b    = blockIdx.x >> 4;
  const int h    = blockIdx.x & 15;
  const size_t base = (size_t)b * SEQ * D_MODEL + (size_t)h * DK;

  // Async-load Q and K tiles (each 128x128 f16) straight into LDS.
#pragma unroll
  for (int i = 0; i < 8; ++i) {
    const int c  = tid + 256 * i;
    const int r  = c >> 4;
    const int c8 = c & 15;
    async_b128(&smA[r * 128 + c8 * 8], &Q[base + (size_t)r * D_MODEL + c8 * 8]);
    async_b128(&smB[r * 128 + c8 * 8],
               &Kmat[base + (size_t)r * D_MODEL + c8 * 8]);
  }
  wait_async0();
  __syncthreads();

  const int row0  = wave * 16;
  const int ihalf = row0 + ((lane >> 4) ? 8 : 0);

  v8f acc[8];
#pragma unroll
  for (int nt = 0; nt < 8; ++nt) {
    v8f z = {0.f, 0.f, 0.f, 0.f, 0.f, 0.f, 0.f, 0.f};
    acc[nt] = z;
  }
#pragma unroll
  for (int kk = 0; kk < 4; ++kk) {
    const int k0 = kk * 32;
    const v16h a = load_frag_a(&smA[row0 * 128 + k0], 128, lane);
#pragma unroll
    for (int nt = 0; nt < 8; ++nt) {
      const v16h bb = load_frag_b(&smB[(nt * 16) * 128 + k0], 128, lane);
      acc[nt] = wmma_f16(a, bb, acc[nt]);
    }
  }

  // scale + relative-position bias
  const float scale = 0.08838834764831845f;  // 1/sqrt(128)
#pragma unroll
  for (int nt = 0; nt < 8; ++nt) {
    const int j = nt * 16 + (lane & 15);
#pragma unroll
    for (int r = 0; r < 8; ++r) {
      const int i = ihalf + r;
      acc[nt][r] = acc[nt][r] * scale +
                   rel[(size_t)(((b - i) + REL_T) & (REL_T - 1)) * DK + j];
    }
  }

  // softmax over j: per row, values live in 8 accum slots x 16 lanes
#pragma unroll
  for (int r = 0; r < 8; ++r) {
    float m = -3.0e38f;
#pragma unroll
    for (int nt = 0; nt < 8; ++nt) m = fmaxf(m, acc[nt][r]);
    for (int off = 8; off; off >>= 1) m = fmaxf(m, __shfl_xor(m, off, 32));
    float s = 0.f;
#pragma unroll
    for (int nt = 0; nt < 8; ++nt) {
      const float e = __expf(acc[nt][r] - m);
      acc[nt][r] = e;
      s += e;
    }
    for (int off = 8; off; off >>= 1) s += __shfl_xor(s, off, 32);
    const float inv = 1.f / s;
#pragma unroll
    for (int nt = 0; nt < 8; ++nt) acc[nt][r] *= inv;
  }
  __syncthreads();  // all waves done reading Q/K from LDS

  // write attn probs (f16) into smA; load V^T into smB
#pragma unroll
  for (int nt = 0; nt < 8; ++nt) {
    const int j = nt * 16 + (lane & 15);
#pragma unroll
    for (int r = 0; r < 8; ++r) smA[(ihalf + r) * 128 + j] = (f16)acc[nt][r];
  }
#pragma unroll
  for (int i = 0; i < 8; ++i) {
    const int c  = tid + 256 * i;
    const int j  = c >> 4;   // sequence index
    const int c8 = c & 15;   // d chunk
    const uq4 d = *(const uq4*)&V[base + (size_t)j * D_MODEL + c8 * 8];
    const f16* dh = (const f16*)&d;
#pragma unroll
    for (int t = 0; t < 8; ++t) smB[(c8 * 8 + t) * 128 + j] = dh[t];
  }
  __syncthreads();

  // ctx = attn @ V
  v8f acc2[8];
#pragma unroll
  for (int nt = 0; nt < 8; ++nt) {
    v8f z = {0.f, 0.f, 0.f, 0.f, 0.f, 0.f, 0.f, 0.f};
    acc2[nt] = z;
  }
#pragma unroll
  for (int kk = 0; kk < 4; ++kk) {
    const int k0 = kk * 32;
    const v16h a = load_frag_a(&smA[row0 * 128 + k0], 128, lane);
#pragma unroll
    for (int nt = 0; nt < 8; ++nt) {
      const v16h bb = load_frag_b(&smB[(nt * 16) * 128 + k0], 128, lane);
      acc2[nt] = wmma_f16(a, bb, acc2[nt]);
    }
  }
#pragma unroll
  for (int nt = 0; nt < 8; ++nt) {
    const int d = nt * 16 + (lane & 15);
#pragma unroll
    for (int r = 0; r < 8; ++r) {
      const int i = ihalf + r;
      ctx[base + (size_t)i * D_MODEL + d] = (f16)acc2[nt][r];
    }
  }
}

// ---------------------------------------------------------------------------
// out = LayerNorm(x + y) * g + b ; optional f16 copy for the next GEMM.
// ---------------------------------------------------------------------------
__global__ __launch_bounds__(256) void add_ln_kernel(
    const float* __restrict__ x, const float* __restrict__ y,
    const float* __restrict__ g, const float* __restrict__ bta,
    float* __restrict__ outf, f16* __restrict__ outh) {
  __shared__ float red[256];
  const int tid = threadIdx.x;
  const size_t off = (size_t)blockIdx.x * D_MODEL;

  float vals[8];
  float s = 0.f;
#pragma unroll
  for (int i = 0; i < 8; ++i) {
    const int col = tid + 256 * i;
    const float v = x[off + col] + y[off + col];
    vals[i] = v;
    s += v;
  }
  red[tid] = s;
  __syncthreads();
  for (int st = 128; st; st >>= 1) {
    if (tid < st) red[tid] += red[tid + st];
    __syncthreads();
  }
  const float mu = red[0] * (1.f / (float)D_MODEL);
  __syncthreads();

  float q = 0.f;
#pragma unroll
  for (int i = 0; i < 8; ++i) {
    const float d = vals[i] - mu;
    q += d * d;
  }
  red[tid] = q;
  __syncthreads();
  for (int st = 128; st; st >>= 1) {
    if (tid < st) red[tid] += red[tid + st];
    __syncthreads();
  }
  const float rstd = rsqrtf(red[0] * (1.f / (float)D_MODEL) + 1e-5f);

#pragma unroll
  for (int i = 0; i < 8; ++i) {
    const int col = tid + 256 * i;
    const float o = (vals[i] - mu) * rstd * g[col] + bta[col];
    outf[off + col] = o;
    if (outh) outh[off + col] = (f16)o;
  }
}

// ---------------------------------------------------------------------------
// Host orchestration
// ---------------------------------------------------------------------------
static inline char* ws_take(char*& cur, size_t bytes) {
  char* p = cur;
  cur += (bytes + 255) & ~(size_t)255;
  return p;
}

extern "C" void kernel_launch(void* const* d_in, const int* in_sizes, int n_in,
                              void* d_out, int out_size, void* d_ws,
                              size_t ws_size, hipStream_t stream) {
  const float* x     = (const float*)d_in[0];
  const float* Wq    = (const float*)d_in[1];
  const float* bq    = (const float*)d_in[2];
  const float* Wk    = (const float*)d_in[3];
  const float* bk    = (const float*)d_in[4];
  const float* Wv    = (const float*)d_in[5];
  const float* bv    = (const float*)d_in[6];
  const float* Wo    = (const float*)d_in[7];
  const float* bo    = (const float*)d_in[8];
  const float* rel   = (const float*)d_in[9];
  const float* ln1_g = (const float*)d_in[10];
  const float* ln1_b = (const float*)d_in[11];
  const float* W1    = (const float*)d_in[12];
  const float* b1    = (const float*)d_in[13];
  const float* W2    = (const float*)d_in[14];
  const float* b2    = (const float*)d_in[15];
  const float* ln2_g = (const float*)d_in[16];
  const float* ln2_b = (const float*)d_in[17];
  float* out = (float*)d_out;

  const size_t ND = (size_t)NTOK * D_MODEL;
  const size_t NF = (size_t)NTOK * D_FF;
  const size_t DD = (size_t)D_MODEL * D_MODEL;
  const size_t DF = (size_t)D_MODEL * D_FF;

  char* cur = (char*)d_ws;
  f16*  xh  = (f16*)ws_take(cur, ND * 2);
  f16*  Wqt = (f16*)ws_take(cur, DD * 2);
  f16*  Wkt = (f16*)ws_take(cur, DD * 2);
  f16*  Wvt = (f16*)ws_take(cur, DD * 2);
  f16*  Wot = (f16*)ws_take(cur, DD * 2);
  f16*  W1t = (f16*)ws_take(cur, DF * 2);
  f16*  W2t = (f16*)ws_take(cur, DF * 2);
  // Q,K,V,ctx (4 x 64MB) alias the FFN hidden buffer h (NTOK x D_FF f16)
  char* qkv  = ws_take(cur, NF * 2);
  f16*  Qh   = (f16*)qkv;
  f16*  Kh   = Qh + ND;
  f16*  Vh   = Kh + ND;
  f16*  ctxh = Vh + ND;
  f16*  hh   = (f16*)qkv;  // alias (used after attention block is done)
  float* attnf = (float*)ws_take(cur, ND * 4);  // reused as ff buffer
  float* fff   = attnf;
  float* x1f   = (float*)ws_take(cur, ND * 4);
  f16*   x1h   = (f16*)ws_take(cur, ND * 2);

  // 1) convert x -> f16 ; transpose+convert all weights -> [N][K] f16
  cvt_f32_to_f16_kernel<<<4096, 256, 0, stream>>>(x, xh, ND);
  {
    dim3 b(256);
    transpose_cvt_kernel<<<dim3(64, 64), b, 0, stream>>>(Wq, Wqt, D_MODEL, D_MODEL);
    transpose_cvt_kernel<<<dim3(64, 64), b, 0, stream>>>(Wk, Wkt, D_MODEL, D_MODEL);
    transpose_cvt_kernel<<<dim3(64, 64), b, 0, stream>>>(Wv, Wvt, D_MODEL, D_MODEL);
    transpose_cvt_kernel<<<dim3(64, 64), b, 0, stream>>>(Wo, Wot, D_MODEL, D_MODEL);
    transpose_cvt_kernel<<<dim3(256, 64), b, 0, stream>>>(W1, W1t, D_MODEL, D_FF);
    transpose_cvt_kernel<<<dim3(64, 256), b, 0, stream>>>(W2, W2t, D_FF, D_MODEL);
  }

  // 2) Q/K/V projections (bias fused, f16 out; [b,s,h*dk] layout == strided [b,h])
  {
    dim3 g(D_MODEL / 256, NTOK / 128), b(256);
    gemm_wmma_kernel<1><<<g, b, 0, stream>>>(xh, Wqt, bq, nullptr, Qh,
                                             NTOK, D_MODEL, D_MODEL);
    gemm_wmma_kernel<1><<<g, b, 0, stream>>>(xh, Wkt, bk, nullptr, Kh,
                                             NTOK, D_MODEL, D_MODEL);
    gemm_wmma_kernel<1><<<g, b, 0, stream>>>(xh, Wvt, bv, nullptr, Vh,
                                             NTOK, D_MODEL, D_MODEL);
  }

  // 3) fused attention (one block per (b,h))
  attn_kernel<<<NTOK / SEQ * NH, 256, 0, stream>>>(Qh, Kh, Vh, rel, ctxh);

  // 4) output projection (bias fused, f32 out)
  gemm_wmma_kernel<2><<<dim3(D_MODEL / 256, NTOK / 128), 256, 0, stream>>>(
      ctxh, Wot, bo, attnf, nullptr, NTOK, D_MODEL, D_MODEL);

  // 5) x1 = LN(x + attn_out)  (+f16 copy)
  add_ln_kernel<<<NTOK, 256, 0, stream>>>(x, attnf, ln1_g, ln1_b, x1f, x1h);

  // 6) h = gelu(x1 @ W1 + b1)  (f16 out, aliases QKV/ctx space)
  gemm_wmma_kernel<3><<<dim3(D_FF / 256, NTOK / 128), 256, 0, stream>>>(
      x1h, W1t, b1, nullptr, hh, NTOK, D_FF, D_MODEL);

  // 7) ff = h @ W2 + b2  (f32 out, aliases attnf)
  gemm_wmma_kernel<2><<<dim3(D_MODEL / 256, NTOK / 128), 256, 0, stream>>>(
      hh, W2t, b2, fff, nullptr, NTOK, D_MODEL, D_FF);

  // 8) out = LN(x1 + ff)
  add_ln_kernel<<<NTOK, 256, 0, stream>>>(x1f, fff, ln2_g, ln2_b, out, nullptr);
}
